// RegressionWisard_61100204752929
// MI455X (gfx1250) — 8-bit matrix operand order repaired and checked by reading, compiled, not run on hardware
//
#include <hip/hip_runtime.h>
#include <stdint.h>

typedef int          v8i __attribute__((ext_vector_type(8)));
typedef int          v4i __attribute__((ext_vector_type(4)));
typedef unsigned int v4u __attribute__((ext_vector_type(4)));

#define ENTRY_SIZE 8192
#define TUPLE_SIZE 16
#define N_NEURONS  512
#define ADDR_SPACE 65536
#define BATCH      4096

__global__ __launch_bounds__(256) void wisard_predict_kernel(
    const int*   __restrict__ input,
    const int*   __restrict__ tmap,
    const int*   __restrict__ counts,
    const float* __restrict__ sums,
    float*       __restrict__ out)
{
    __shared__ int   lds_bits[ENTRY_SIZE];
    __shared__ float s_resp[8];
    __shared__ int   s_cnt[8];

    const int b    = blockIdx.x;
    const int tid  = threadIdx.x;
    const int lane = tid & 31;
    const int wave = tid >> 5;

    if (wave == 0) {
        uint64_t gaddr = (uint64_t)(uintptr_t)(input + (size_t)b * ENTRY_SIZE);
        uint32_t laddr = (uint32_t)(uintptr_t)(void*)lds_bits;

        v4u g0;
        g0.x = 1u;
        g0.y = laddr;
        g0.z = (uint32_t)gaddr;
        g0.w = ((uint32_t)(gaddr >> 32) & 0x01FFFFFFu) | (2u << 30);

        v8i g1;
        g1[0] = (int)(2u << 16);
        g1[1] = (int)((uint32_t)(ENTRY_SIZE & 0xFFFF) << 16);
        g1[2] = (int)(((uint32_t)ENTRY_SIZE >> 16) | (1u << 16));
        g1[3] = (int)((uint32_t)ENTRY_SIZE << 16);
        g1[4] = 1;
        g1[5] = ENTRY_SIZE;
        g1[6] = 0;
        g1[7] = 0;
        v4i gz4 = {0, 0, 0, 0};
        v8i gz8 = {0, 0, 0, 0, 0, 0, 0, 0};

        __builtin_amdgcn_tensor_load_to_lds(g0, g1, gz4, gz4, gz8, 0);
        __builtin_amdgcn_s_wait_tensorcnt(0);
    }
    __syncthreads();

    const int m = lane & 15;
    const int h = lane >> 4;
    const int base_neuron = wave * 64;
    const int p0 = base_neuron * TUPLE_SIZE + m * 64;

    v8i A;
    #pragma unroll
    for (int v = 0; v < 8; ++v) {
        const int kbase = (v >> 1) * 16 + h * 8 + (v & 1) * 4;
        unsigned int packed = 0;
        #pragma unroll
        for (int j = 0; j < 4; ++j) {
            const int k   = kbase + j;
            const int src = tmap[p0 + k];
            const unsigned int bit = (unsigned int)lds_bits[src] & 1u;
            packed |= bit << (8 * j);
        }
        A[v] = (int)packed;
    }

    const int slot = m & 3;
    v8i Blo, Bhi;
    #pragma unroll
    for (int v = 0; v < 8; ++v) {
        unsigned int lo = 0, hi = 0;
        #pragma unroll
        for (int j = 0; j < 4; ++j) {
            const int k = (v >> 1) * 16 + h * 8 + (v & 1) * 4 + j;
            const int g = k >> 4;
            const int t = k & 15;
            if (g == slot) {
                if (t >= 8) lo |= (1u << (15 - t)) << (8 * j);
                else        hi |= (1u << (7  - t)) << (8 * j);
            }
        }
        Blo[v] = (int)lo;
        Bhi[v] = (int)hi;
    }

    const v8i z = {0, 0, 0, 0, 0, 0, 0, 0};
    v8i Dlo = __builtin_amdgcn_wmma_i32_16x16x64_iu8(false, A, false, Blo, z, false, false);
    v8i Dhi = __builtin_amdgcn_wmma_i32_16x16x64_iu8(false, A, false, Bhi, z, false, false);

    float resp = 0.0f;
    int   cnt  = 0;
    const int q = m >> 2;
    #pragma unroll
    for (int e = 0; e < 2; ++e) {
        const int r      = 2 * q + e;
        const int addr   = Dlo[r] + (Dhi[r] << 8);
        const int neuron = base_neuron + 4 * (r + 8 * h) + slot;
        const size_t idx = (size_t)neuron * ADDR_SPACE + (unsigned)addr;
        const int c = counts[idx];
        if (c > 0) {
            resp += sums[idx];
            cnt  += c;
        }
    }

    #pragma unroll
    for (int off = 16; off > 0; off >>= 1) {
        resp += __shfl_down(resp, off, 32);
        cnt  += __shfl_down(cnt,  off, 32);
    }
    if (lane == 0) { s_resp[wave] = resp; s_cnt[wave] = cnt; }
    __syncthreads();

    if (tid == 0) {
        float R = 0.0f;
        int   C = 0;
        #pragma unroll
        for (int w = 0; w < 8; ++w) { R += s_resp[w]; C += s_cnt[w]; }
        out[b] = (C > 0) ? (R / (float)C) : 0.0f;
    }
}

extern "C" void kernel_launch(void* const* d_in, const int* in_sizes, int n_in,
                              void* d_out, int out_size, void* d_ws, size_t ws_size,
                              hipStream_t stream) {
    const int*   input  = (const int*)d_in[0];
    const int*   tmap   = (const int*)d_in[1];
    const int*   counts = (const int*)d_in[2];
    const float* sums   = (const float*)d_in[3];
    float*       out    = (float*)d_out;

    wisard_predict_kernel<<<BATCH, 256, 0, stream>>>(input, tmap, counts, sums, out);
}
